// SelfAttention_v1_90855738180064
// MI455X (gfx1250) — matrix operand order repaired and checked by reading, not compiled, not run on hardware
//
#include <hip/hip_runtime.h>
#include <hip/hip_bf16.h>

// ---------------------------------------------------------------------------
// Self-attention for MI455X (gfx1250, wave32, WMMA).
//   Q = x Wq ; K = x Wk ; V = x Wv            (bf16 outputs in d_ws)
//   S = (Q K^T) / sqrt(d)                     (f32 in d_ws) -- async-copy GEMM
//   S = softmax_rows(S)                       (in place, f32)
//   out = S V                                 (f32, d_out)
// All matmuls use V_WMMA_F32_16X16X32_BF16 with f32 accumulation. The score
// GEMM uses GLOBAL_LOAD_ASYNC_TO_LDS_B128 with double-buffered LDS tiles
// (ASYNCcnt-tracked pipeline); the f32-input GEMMs convert to bf16 in VGPRs
// while staging.
// Workspace requirement: 3*4096*1024*2 + 4096*4096*4 = 88 MB.
// ---------------------------------------------------------------------------

typedef __attribute__((ext_vector_type(16))) __bf16 v16bf;
typedef __attribute__((ext_vector_type(8)))  __bf16 v8bf;
typedef __attribute__((ext_vector_type(8)))  float  v8f;
typedef __attribute__((ext_vector_type(4)))  float  v4f;
typedef __attribute__((ext_vector_type(4)))  int    v4i;

#define SEQ   4096
#define D_IN  1024
#define D_OUT 1024

constexpr int TILE = 128;   // block tile (M and N)
constexpr int TK   = 32;    // K step == WMMA K for bf16
constexpr int LDT  = 34;    // LDS pitch, generic kernel (68B = 17 words, odd -> conflict-free strides)
constexpr int LDA2 = 40;    // LDS pitch, async kernel (80B rows, 16B-aligned chunks for b128)

// Async global->LDS builtins (gfx1250). Guarded so the file compiles on the
// host pass and on toolchains lacking the builtins (falls back to sync stores).
#if defined(__AMDGCN__) && __has_builtin(__builtin_amdgcn_global_load_async_to_lds_b128) && \
    __has_builtin(__builtin_amdgcn_s_wait_asynccnt)
#define HAVE_ASYNC_DEV 1
#else
#define HAVE_ASYNC_DEV 0
#endif

// copy 8 bf16 (16 bytes) from global to LDS; async when available
__device__ inline void async_cp16B(const __bf16* __restrict__ g, __bf16* __restrict__ l) {
#if HAVE_ASYNC_DEV
  __builtin_amdgcn_global_load_async_to_lds_b128(
      (__attribute__((address_space(1))) v4i*)g,
      (__attribute__((address_space(3))) v4i*)l, 0, 0);
#else
#pragma unroll
  for (int j = 0; j < 8; ++j) l[j] = g[j];
#endif
}

template <int N>
__device__ inline void wait_asynccnt() {
#if HAVE_ASYNC_DEV
  __builtin_amdgcn_s_wait_asynccnt(N);
#endif
}

// ---- vectorized 16-element global loads (alignment: offsets are multiples of 16 elems) ----
__device__ inline void load16(const float* __restrict__ g, float* t) {
#pragma unroll
  for (int c = 0; c < 4; ++c) {
    v4f v = reinterpret_cast<const v4f*>(g)[c];          // global_load_b128
#pragma unroll
    for (int j = 0; j < 4; ++j) t[4 * c + j] = v[j];
  }
}
__device__ inline void load16(const __bf16* __restrict__ g, float* t) {
#pragma unroll
  for (int c = 0; c < 2; ++c) {
    v8bf v = reinterpret_cast<const v8bf*>(g)[c];        // global_load_b128
#pragma unroll
    for (int j = 0; j < 8; ++j) t[8 * c + j] = (float)v[j];
  }
}

// ---------------------------------------------------------------------------
// Generic: C[M,N] = alpha * A[M,K] * (TRANSB ? B[N,K]^T : B[K,N])
// 256 threads = 8 waves; block tile 128x128; per-wave 64x32 (4x2 WMMA tiles).
// ---------------------------------------------------------------------------
template <typename TA, typename TB, bool TRANSB, typename TOUT>
__global__ __launch_bounds__(256) void gemm_wmma(
    const TA* __restrict__ A, const TB* __restrict__ B, TOUT* __restrict__ C,
    int M, int N, int K, float alpha) {
  __shared__ __bf16 lA[TILE * LDT];
  __shared__ __bf16 lB[TILE * LDT];

  const int tid  = threadIdx.x;
  const int lane = tid & 31;
  const int wave = tid >> 5;
  const int wm   = wave & 1;        // 0..1 : 64-row slab
  const int wn   = wave >> 1;       // 0..3 : 32-col slab
  const int m0   = blockIdx.y * TILE;
  const int n0   = blockIdx.x * TILE;

  // staging assignments: 256 threads cover 128 rows x 32 cols, 16 elems/thread
  const int srow  = tid >> 1;             // 0..127
  const int shalf = (tid & 1) * 16;       // 0 or 16
  const int bk    = tid >> 3;             // 0..31  (non-transposed B)
  const int bng   = (tid & 7) * 16;       // 0..112

  v8f acc[4][2] = {};

  for (int kb = 0; kb < K; kb += TK) {
    // ---- stage A tile: lA[m][k] = bf16(A[m0+m][kb+k]) ----
    {
      const TA* g = A + (size_t)(m0 + srow) * K + kb + shalf;
      if (kb + TK < K) __builtin_prefetch(g + TK, 0, 1);  // global_prefetch_b8
      float t[16];
      load16(g, t);
      __bf16* l = lA + srow * LDT + shalf;
#pragma unroll
      for (int j = 0; j < 16; ++j) l[j] = (__bf16)t[j];
    }
    // ---- stage B tile as lB[n][k] ----
    if constexpr (TRANSB) {
      const TB* g = B + (size_t)(n0 + srow) * K + kb + shalf;
      if (kb + TK < K) __builtin_prefetch(g + TK, 0, 1);
      float t[16];
      load16(g, t);
      __bf16* l = lB + srow * LDT + shalf;
#pragma unroll
      for (int j = 0; j < 16; ++j) l[j] = (__bf16)t[j];
    } else {
      const TB* g = B + (size_t)(kb + bk) * N + n0 + bng;
      if (kb + TK < K) __builtin_prefetch(g + (size_t)TK * N, 0, 1);
      float t[16];
      load16(g, t);
#pragma unroll
      for (int j = 0; j < 16; ++j) lB[(bng + j) * LDT + bk] = (__bf16)t[j];
    }
    __syncthreads();

    // ---- fragments per ISA bf16 WMMA layouts ----
    v16bf af[4];
#pragma unroll
    for (int i = 0; i < 4; ++i) {
      const __bf16* p = lA + (wm * 64 + i * 16 + (lane & 15)) * LDT + ((lane >> 4) * 8);
#pragma unroll
      for (int e = 0; e < 8; ++e) { af[i][e] = p[e]; af[i][e + 8] = p[e + 16]; }
    }
    v16bf bfr[2];
#pragma unroll
    for (int j = 0; j < 2; ++j) {
      const __bf16* p = lB + (wn * 32 + j * 16 + (lane & 15)) * LDT + ((lane >> 4) * 8);
#pragma unroll
      for (int e = 0; e < 8; ++e) { bfr[j][e] = p[e]; bfr[j][e + 8] = p[e + 16]; }
    }

#pragma unroll
    for (int i = 0; i < 4; ++i)
#pragma unroll
      for (int j = 0; j < 2; ++j)
        acc[i][j] = __builtin_amdgcn_wmma_f32_16x16x32_bf16(
            false, af[i], false, bfr[j], (short)0, acc[i][j], false, false);

    __syncthreads();
  }

  const int rb = (lane >> 4) * 8;
  const int nc = lane & 15;
#pragma unroll
  for (int i = 0; i < 4; ++i)
#pragma unroll
    for (int j = 0; j < 2; ++j) {
      size_t base = (size_t)(m0 + wm * 64 + i * 16 + rb) * N + (n0 + wn * 32 + j * 16 + nc);
#pragma unroll
      for (int r = 0; r < 8; ++r)
        C[base + (size_t)r * N] = static_cast<TOUT>(alpha * acc[i][j][r]);
    }
}

// ---------------------------------------------------------------------------
// Specialized score GEMM: S[M,N] = alpha * Q[M,K] * Kmat[N,K]^T  (bf16 inputs)
// Double-buffered LDS fed by GLOBAL_LOAD_ASYNC_TO_LDS_B128; copy of tile i+1
// overlaps the 8 WMMAs of tile i. ASYNCcnt pipeline: 4 copies in flight per
// thread per tile; s_wait_asynccnt 4 == previous tile fully landed.
// ---------------------------------------------------------------------------
__global__ __launch_bounds__(256) void gemm_score(
    const __bf16* __restrict__ Q, const __bf16* __restrict__ Kmat,
    float* __restrict__ S, int M, int N, int K, float alpha) {
  __shared__ __bf16 lA[2][TILE * LDA2];
  __shared__ __bf16 lB[2][TILE * LDA2];

  const int tid  = threadIdx.x;
  const int lane = tid & 31;
  const int wave = tid >> 5;
  const int wm   = wave & 1;
  const int wn   = wave >> 1;
  const int m0   = blockIdx.y * TILE;
  const int n0   = blockIdx.x * TILE;

  const int srow  = tid >> 1;            // 0..127
  const int shalf = (tid & 1) * 16;      // 0 or 16

  const __bf16* ga = Q    + (size_t)(m0 + srow) * K + shalf;
  const __bf16* gb = Kmat + (size_t)(n0 + srow) * K + shalf;
  __bf16* la = &lA[0][srow * LDA2 + shalf];
  __bf16* lb = &lB[0][srow * LDA2 + shalf];
  constexpr int BUFSTRIDE = TILE * LDA2;

  auto issue = [&](int kb, int buf) {
    async_cp16B(ga + kb,     la + buf * BUFSTRIDE);
    async_cp16B(ga + kb + 8, la + buf * BUFSTRIDE + 8);
    async_cp16B(gb + kb,     lb + buf * BUFSTRIDE);
    async_cp16B(gb + kb + 8, lb + buf * BUFSTRIDE + 8);
  };

  v8f acc[4][2] = {};
  const int T = K / TK;

  issue(0, 0);
  for (int i = 0; i < T; ++i) {
    if (i + 1 < T) {
      issue((i + 1) * TK, (i + 1) & 1);  // prefetch next tile into other buffer
      wait_asynccnt<4>();                // tile i complete (in-order async)
    } else {
      wait_asynccnt<0>();
    }
    __syncthreads();

    const __bf16* bufA = lA[i & 1];
    const __bf16* bufB = lB[i & 1];

    v16bf af[4];
#pragma unroll
    for (int ii = 0; ii < 4; ++ii) {
      const __bf16* p = bufA + (wm * 64 + ii * 16 + (lane & 15)) * LDA2 + ((lane >> 4) * 8);
#pragma unroll
      for (int e = 0; e < 8; ++e) { af[ii][e] = p[e]; af[ii][e + 8] = p[e + 16]; }
    }
    v16bf bfr[2];
#pragma unroll
    for (int j = 0; j < 2; ++j) {
      const __bf16* p = bufB + (wn * 32 + j * 16 + (lane & 15)) * LDA2 + ((lane >> 4) * 8);
#pragma unroll
      for (int e = 0; e < 8; ++e) { bfr[j][e] = p[e]; bfr[j][e + 8] = p[e + 16]; }
    }

#pragma unroll
    for (int ii = 0; ii < 4; ++ii)
#pragma unroll
      for (int j = 0; j < 2; ++j)
        acc[ii][j] = __builtin_amdgcn_wmma_f32_16x16x32_bf16(
            false, af[ii], false, bfr[j], (short)0, acc[ii][j], false, false);

    __syncthreads();  // everyone done reading buf before it is re-filled
  }

  const int rb = (lane >> 4) * 8;
  const int nc = lane & 15;
#pragma unroll
  for (int i = 0; i < 4; ++i)
#pragma unroll
    for (int j = 0; j < 2; ++j) {
      size_t base = (size_t)(m0 + wm * 64 + i * 16 + rb) * N + (n0 + wn * 32 + j * 16 + nc);
#pragma unroll
      for (int r = 0; r < 8; ++r)
        S[base + (size_t)r * N] = alpha * acc[i][j][r];
    }
}

// ---------------------------------------------------------------------------
// In-place row softmax over S[rows][n], n == 4096, one 256-thread block per row.
// ---------------------------------------------------------------------------
__global__ __launch_bounds__(256) void softmax_rows(float* __restrict__ S, int n) {
  float* p = S + (size_t)blockIdx.x * n;
  const int tid = threadIdx.x, lane = tid & 31, w = tid >> 5;
  __shared__ float red[8];

  float v[16];
  float m = -3.402823466e38f;
#pragma unroll
  for (int j = 0; j < 16; ++j) {
    v[j] = p[tid + j * 256];
    m = fmaxf(m, v[j]);
  }
#pragma unroll
  for (int o = 16; o > 0; o >>= 1) m = fmaxf(m, __shfl_xor(m, o, 32));
  if (lane == 0) red[w] = m;
  __syncthreads();
  float bm = red[0];
#pragma unroll
  for (int i = 1; i < 8; ++i) bm = fmaxf(bm, red[i]);

  float s = 0.f;
#pragma unroll
  for (int j = 0; j < 16; ++j) {
    v[j] = __expf(v[j] - bm);
    s += v[j];
  }
#pragma unroll
  for (int o = 16; o > 0; o >>= 1) s += __shfl_xor(s, o, 32);
  __syncthreads();                 // done reading red[] for max
  if (lane == 0) red[w] = s;
  __syncthreads();
  float bs = 0.f;
#pragma unroll
  for (int i = 0; i < 8; ++i) bs += red[i];
  const float inv = 1.0f / bs;
#pragma unroll
  for (int j = 0; j < 16; ++j) p[tid + j * 256] = v[j] * inv;
}

// ---------------------------------------------------------------------------
extern "C" void kernel_launch(void* const* d_in, const int* in_sizes, int n_in,
                              void* d_out, int out_size, void* d_ws, size_t ws_size,
                              hipStream_t stream) {
  const float* x  = (const float*)d_in[0];
  const float* Wq = (const float*)d_in[1];
  const float* Wk = (const float*)d_in[2];
  const float* Wv = (const float*)d_in[3];
  float* out = (float*)d_out;

  // workspace layout: Qb | Kb | Vb (bf16, 8MB each) | S (f32, 64MB)
  __bf16* Qb = (__bf16*)d_ws;
  __bf16* Kb = Qb + (size_t)SEQ * D_OUT;
  __bf16* Vb = Kb + (size_t)SEQ * D_OUT;
  float*  S  = (float*)(Vb + (size_t)SEQ * D_OUT);

  dim3 blk(256);
  dim3 gProj(D_OUT / TILE, SEQ / TILE);   // (8, 32)
  dim3 gScore(SEQ / TILE, SEQ / TILE);    // (32, 32)

  // Q/K/V projections: f32 inputs, bf16 outputs
  gemm_wmma<float, float, false, __bf16><<<gProj, blk, 0, stream>>>(x, Wq, Qb, SEQ, D_OUT, D_IN, 1.0f);
  gemm_wmma<float, float, false, __bf16><<<gProj, blk, 0, stream>>>(x, Wk, Kb, SEQ, D_OUT, D_IN, 1.0f);
  gemm_wmma<float, float, false, __bf16><<<gProj, blk, 0, stream>>>(x, Wv, Vb, SEQ, D_OUT, D_IN, 1.0f);

  // S = Q K^T / sqrt(1024): async-copy double-buffered WMMA GEMM
  gemm_score<<<gScore, blk, 0, stream>>>(Qb, Kb, S, SEQ, SEQ, D_OUT, 0.03125f);

  // row softmax in place
  softmax_rows<<<SEQ, blk, 0, stream>>>(S, SEQ);

  // out = P V  (P converted to bf16 in staging)
  gemm_wmma<float, __bf16, false, float><<<gProj, blk, 0, stream>>>(S, Vb, out, SEQ, D_OUT, SEQ, 1.0f);
}
